// Block_graph_42666205118973
// MI455X (gfx1250) — compile-verified, matrix operand order checked by reading, not run on hardware
//
#include <hip/hip_runtime.h>
#include <hip/hip_bf16.h>

// ---------------------------------------------------------------------------
// Problem constants (from reference)
// ---------------------------------------------------------------------------
#define S_LEN 2049
#define H_DIM 768
#define HEADS 12
#define DH 64
#define MLP_DIM 3072
#define N_NODES 2048
#define E_EDGES 32768
#define H2_DIM 1536
#define H3_DIM 2304

typedef __attribute__((ext_vector_type(16))) _Float16 v16h;
typedef __attribute__((ext_vector_type(8)))  float    v8f;

// ---------------------------------------------------------------------------
// Generic tiled GEMM with f16 WMMA (fp32 accumulate).
//   C[b] = act( alpha * A[b] @ B[b] + bias ) + res[b]
// A: M x K  (row-major, leading dim lda), batch stride sAb (elements)
// B: bT==0 -> K x N row-major (weights);  bT==1 -> N x K row-major (e.g. K^T)
// Block tile 128(M) x 64(N) x 32(K); 256 threads = 8 waves arranged 4x2;
// wave tile 32x32 = 4 WMMA accumulators (2 A frags x 2 B frags).
// Interior tiles use float4 global loads + 8B LDS stores; edges fall back to
// guarded scalar staging.
// ---------------------------------------------------------------------------
#define TM 128
#define TN 64
#define TK 32
#define KSTR 36   // padded LDS k-stride (halves); even -> 4B/8B aligned pairs

__global__ __launch_bounds__(256)
void gemm_f16_wmma(const float* __restrict__ A, long long sAb, int lda,
                   const float* __restrict__ B, long long sBb, int ldb, int bT,
                   const float* __restrict__ bias,
                   const float* __restrict__ res, long long sRb, int ldres,
                   float* __restrict__ C, long long sCb, int ldc,
                   int M, int N, int K, float alpha, int act)
{
    __shared__ __align__(16) _Float16 As[TM * KSTR];   // 9.2 KB
    __shared__ __align__(16) _Float16 Bs[TN * KSTR];   // 4.6 KB

    const int bz = blockIdx.z;
    A += bz * sAb;
    B += bz * sBb;
    C += bz * sCb;
    if (res) res += bz * sRb;

    const int m0  = blockIdx.y * TM;
    const int n0  = blockIdx.x * TN;
    const int tid = threadIdx.x;
    const int wave = tid >> 5;
    const int lane = tid & 31;
    const int wm = wave >> 1;          // 0..3 -> 32-row M sub-tile
    const int wn = wave & 1;           // 0..1 -> 32-col N sub-tile

    v8f acc[2][2] = {};

    const int lrow = lane & 15;
    const int kb   = (lane >> 4) << 3; // 0 or 8 (16-bit fragment K base)
    const bool mful = (m0 + TM <= M);
    const bool nful = (n0 + TN <= N);
    const bool aVec = ((lda & 3) == 0);
    const bool bVec = ((ldb & 3) == 0);

    for (int k0 = 0; k0 < K; k0 += TK) {
        const bool kful = (k0 + TK <= K);

        // -------- prefetch next K tile (global_prefetch_b8) --------
        if (k0 + TK < K) {
            __builtin_prefetch(A + (long long)(m0 + (tid & 127)) * lda + (k0 + TK), 0, 3);
            const float* bp = bT
                ? (B + (long long)(n0 + (tid & 63)) * ldb + (k0 + TK))
                : (B + (long long)(k0 + TK + (tid & 31)) * ldb + n0);
            __builtin_prefetch(bp, 0, 3);
        }

        // -------- stage A tile (TM x TK) --------
        if (mful && kful && aVec) {
#pragma unroll
            for (int q = 0; q < 4; ++q) {
                const int u = q * 256 + tid;      // float4 unit, 0..1023
                const int r = u >> 3;             // row 0..127
                const int c = (u & 7) * 4;        // k offset 0..28
                const float4 v = *(const float4*)(A + (long long)(m0 + r) * lda + (k0 + c));
                union { _Float16 h[4]; unsigned long long u64; } t;
                t.h[0] = (_Float16)v.x; t.h[1] = (_Float16)v.y;
                t.h[2] = (_Float16)v.z; t.h[3] = (_Float16)v.w;
                *(unsigned long long*)&As[r * KSTR + c] = t.u64;
            }
        } else {
            for (int i = tid; i < TM * TK; i += 256) {
                const int r = i >> 5, c = i & 31;
                const int gm = m0 + r, gk = k0 + c;
                float v = (gm < M && gk < K) ? A[(long long)gm * lda + gk] : 0.f;
                As[r * KSTR + c] = (_Float16)v;
            }
        }

        // -------- stage B tile as Bs[n][k] --------
        if (nful && kful && bVec) {
            if (bT) {  // B is N x K row-major: float4 along k
#pragma unroll
                for (int q = 0; q < 2; ++q) {
                    const int u = q * 256 + tid;  // 0..511
                    const int n = u >> 3;         // 0..63
                    const int c = (u & 7) * 4;
                    const float4 v = *(const float4*)(B + (long long)(n0 + n) * ldb + (k0 + c));
                    union { _Float16 h[4]; unsigned long long u64; } t;
                    t.h[0] = (_Float16)v.x; t.h[1] = (_Float16)v.y;
                    t.h[2] = (_Float16)v.z; t.h[3] = (_Float16)v.w;
                    *(unsigned long long*)&Bs[n * KSTR + c] = t.u64;
                }
            } else {   // B is K x N row-major: float4 along n, transpose to LDS
#pragma unroll
                for (int q = 0; q < 2; ++q) {
                    const int u = q * 256 + tid;  // 0..511
                    const int k = u >> 4;         // 0..31
                    const int n = (u & 15) * 4;   // 0..60
                    const float4 v = *(const float4*)(B + (long long)(k0 + k) * ldb + (n0 + n));
                    Bs[(n + 0) * KSTR + k] = (_Float16)v.x;
                    Bs[(n + 1) * KSTR + k] = (_Float16)v.y;
                    Bs[(n + 2) * KSTR + k] = (_Float16)v.z;
                    Bs[(n + 3) * KSTR + k] = (_Float16)v.w;
                }
            }
        } else {
            for (int i = tid; i < TN * TK; i += 256) {
                const int n = i >> 5, c = i & 31;
                const int gn = n0 + n, gk = k0 + c;
                float v = 0.f;
                if (gn < N && gk < K)
                    v = bT ? B[(long long)gn * ldb + gk] : B[(long long)gk * ldb + gn];
                Bs[n * KSTR + c] = (_Float16)v;
            }
        }
        __syncthreads();

        // -------- fragments (ISA 16-bit 16x32 layout) + 4 WMMAs --------
        union frag_t { v16h v; unsigned u[8]; } fa[2], fb[2];
        const _Float16* ar0 = &As[(wm * 32 +      lrow) * KSTR];
        const _Float16* ar1 = &As[(wm * 32 + 16 + lrow) * KSTR];
        const _Float16* br0 = &Bs[(wn * 32 +      lrow) * KSTR];
        const _Float16* br1 = &Bs[(wn * 32 + 16 + lrow) * KSTR];
#pragma unroll
        for (int j = 0; j < 8; ++j) {
            const int kk = (j < 4) ? (kb + 2 * j) : (16 + kb + 2 * (j - 4));
            fa[0].u[j] = *(const unsigned*)(ar0 + kk);
            fa[1].u[j] = *(const unsigned*)(ar1 + kk);
            fb[0].u[j] = *(const unsigned*)(br0 + kk);
            fb[1].u[j] = *(const unsigned*)(br1 + kk);
        }
        acc[0][0] = __builtin_amdgcn_wmma_f32_16x16x32_f16(false, fa[0].v, false, fb[0].v,
                                                           (short)0, acc[0][0], false, false);
        acc[0][1] = __builtin_amdgcn_wmma_f32_16x16x32_f16(false, fa[0].v, false, fb[1].v,
                                                           (short)0, acc[0][1], false, false);
        acc[1][0] = __builtin_amdgcn_wmma_f32_16x16x32_f16(false, fa[1].v, false, fb[0].v,
                                                           (short)0, acc[1][0], false, false);
        acc[1][1] = __builtin_amdgcn_wmma_f32_16x16x32_f16(false, fa[1].v, false, fb[1].v,
                                                           (short)0, acc[1][1], false, false);
        __syncthreads();
    }

    // -------- epilogue: C/D layout: lanes 0-15 M=r, lanes 16-31 M=r+8 --------
    const int mo = (lane >> 4) << 3;
    const int ln = lane & 15;
#pragma unroll
    for (int a = 0; a < 2; ++a) {
#pragma unroll
        for (int r = 0; r < 8; ++r) {
            const int gm = m0 + wm * 32 + a * 16 + r + mo;
            if (gm >= M) continue;
#pragma unroll
            for (int b = 0; b < 2; ++b) {
                const int gn = n0 + wn * 32 + b * 16 + ln;
                if (gn >= N) continue;
                float v = acc[a][b][r] * alpha;
                if (bias) v += bias[gn];
                if (act == 1)      v = 0.5f * v * (1.0f + erff(v * 0.70710678118f)); // exact GELU
                else if (act == 2) v = v > 0.f ? v : 0.f;
                if (res) v += res[(long long)gm * ldres + gn];
                C[(long long)gm * ldc + gn] = v;
            }
        }
    }
}

// ---------------------------------------------------------------------------
// LayerNorm over `cols` per row; optional ReLU then optional residual add.
//   out[row] = maybe_relu( (x-mu)*rsqrt(var+eps)*g + b ) + res[row]
// ---------------------------------------------------------------------------
__global__ __launch_bounds__(256)
void ln_kernel(const float* __restrict__ in, int ldin,
               const float* __restrict__ g, const float* __restrict__ b,
               const float* __restrict__ res, int ldres,
               float* __restrict__ out, int ldout,
               int cols, float eps, int dorelu)
{
    const int row = blockIdx.x;
    const float* x = in + (long long)row * ldin;
    __shared__ float red[256];
    const int tid = threadIdx.x;

    float s = 0.f;
    for (int c = tid; c < cols; c += 256) s += x[c];
    red[tid] = s; __syncthreads();
    for (int off = 128; off > 0; off >>= 1) { if (tid < off) red[tid] += red[tid + off]; __syncthreads(); }
    const float mu = red[0] / cols;
    __syncthreads();

    float v = 0.f;
    for (int c = tid; c < cols; c += 256) { const float d = x[c] - mu; v += d * d; }
    red[tid] = v; __syncthreads();
    for (int off = 128; off > 0; off >>= 1) { if (tid < off) red[tid] += red[tid + off]; __syncthreads(); }
    const float rstd = rsqrtf(red[0] / cols + eps);

    for (int c = tid; c < cols; c += 256) {
        float y = (x[c] - mu) * rstd * g[c] + b[c];
        if (dorelu) y = y > 0.f ? y : 0.f;
        if (res) y += res[(long long)row * ldres + c];
        out[(long long)row * ldout + c] = y;
    }
}

// ---------------------------------------------------------------------------
// In-place row softmax (attention weights), row length S.
// ---------------------------------------------------------------------------
__global__ __launch_bounds__(256)
void softmax_rows(float* __restrict__ w, int cols)
{
    float* p = w + (long long)blockIdx.x * cols;
    __shared__ float red[256];
    const int tid = threadIdx.x;

    float m = -3.402823466e+38f;
    for (int c = tid; c < cols; c += 256) m = fmaxf(m, p[c]);
    red[tid] = m; __syncthreads();
    for (int off = 128; off > 0; off >>= 1) { if (tid < off) red[tid] = fmaxf(red[tid], red[tid + off]); __syncthreads(); }
    const float rmax = red[0];
    __syncthreads();

    float s = 0.f;
    for (int c = tid; c < cols; c += 256) { const float e = expf(p[c] - rmax); p[c] = e; s += e; }
    red[tid] = s; __syncthreads();
    for (int off = 128; off > 0; off >>= 1) { if (tid < off) red[tid] += red[tid + off]; __syncthreads(); }
    const float inv = 1.0f / red[0];
    for (int c = tid; c < cols; c += 256) p[c] *= inv;
}

// ---------------------------------------------------------------------------
// GENConv softmax aggregation via atomics.
// ---------------------------------------------------------------------------
__device__ __forceinline__ unsigned enc_f(float f) {
    unsigned u = __float_as_uint(f);
    return (u & 0x80000000u) ? ~u : (u | 0x80000000u);   // order-preserving
}
__device__ __forceinline__ float dec_f(unsigned k) {
    return __uint_as_float((k & 0x80000000u) ? (k ^ 0x80000000u) : ~k);
}

__global__ __launch_bounds__(256)
void edge_max_kernel(const float* __restrict__ xin, int ldin,
                     const int* __restrict__ src, const int* __restrict__ dst,
                     unsigned* __restrict__ keys,
                     const float* __restrict__ tarr, int layer)
{
    const int idx = blockIdx.x * 256 + threadIdx.x;
    if (idx >= E_EDGES * H_DIM) return;
    const int e = idx / H_DIM, c = idx - e * H_DIM;
    float m = xin[(long long)src[e] * ldin + c];
    m = fmaxf(m, 0.f) + 1e-7f;                            // relu + GEN_EPS
    atomicMax(&keys[(long long)dst[e] * H_DIM + c], enc_f(m * tarr[layer]));
}

__global__ __launch_bounds__(256)
void edge_sum_kernel(const float* __restrict__ xin, int ldin,
                     const int* __restrict__ src, const int* __restrict__ dst,
                     const unsigned* __restrict__ keys,
                     float* __restrict__ den, float* __restrict__ num,
                     const float* __restrict__ tarr, int layer)
{
    const int idx = blockIdx.x * 256 + threadIdx.x;
    if (idx >= E_EDGES * H_DIM) return;
    const int e = idx / H_DIM, c = idx - e * H_DIM;
    float m = xin[(long long)src[e] * ldin + c];
    m = fmaxf(m, 0.f) + 1e-7f;
    const float st = m * tarr[layer];
    const long long di = (long long)dst[e] * H_DIM + c;
    const unsigned k = keys[di];
    const float mm = (k == 0u) ? 0.f : dec_f(k);          // empty segment -> 0
    const float a = expf(st - mm);
    atomicAdd(&den[di], a);
    atomicAdd(&num[di], a * m);
}

__global__ __launch_bounds__(256)
void agg_kernel(const float* __restrict__ num, const float* __restrict__ den,
                const float* __restrict__ xin, int ldin,
                float* __restrict__ out)
{
    const int idx = blockIdx.x * 256 + threadIdx.x;
    if (idx >= N_NODES * H_DIM) return;
    const int n = idx / H_DIM, c = idx - n * H_DIM;
    out[idx] = num[idx] / (den[idx] + 1e-16f) + xin[(long long)n * ldin + c];
}

// Column mean over N rows, added into xout (CLS row of output x).
__global__ __launch_bounds__(256)
void mean_add_kernel(const float* __restrict__ g, float* __restrict__ xout)
{
    const int c = blockIdx.x;                 // channel
    __shared__ float red[256];
    const int tid = threadIdx.x;
    float s = 0.f;
    for (int r = tid; r < N_NODES; r += 256) s += g[(long long)r * H_DIM + c];
    red[tid] = s; __syncthreads();
    for (int off = 128; off > 0; off >>= 1) { if (tid < off) red[tid] += red[tid + off]; __syncthreads(); }
    if (tid == 0) xout[c] += red[0] * (1.0f / N_NODES);
}

// ---------------------------------------------------------------------------
// Host orchestration
// ---------------------------------------------------------------------------
static inline dim3 gemm_grid(int M, int N, int batch) {
    return dim3((N + TN - 1) / TN, (M + TM - 1) / TM, batch);
}

extern "C" void kernel_launch(void* const* d_in, const int* in_sizes, int n_in,
                              void* d_out, int out_size, void* d_ws, size_t ws_size,
                              hipStream_t stream)
{
    (void)in_sizes; (void)n_in; (void)out_size; (void)ws_size;

    const float* x      = (const float*)d_in[0];
    const int*   eidx   = (const int*)d_in[1];
    const int*   src    = eidx;
    const int*   dst    = eidx + E_EDGES;
    const float* ln1_g  = (const float*)d_in[2];
    const float* ln1_b  = (const float*)d_in[3];
    const float* ln2_g  = (const float*)d_in[4];
    const float* ln2_b  = (const float*)d_in[5];
    const float* wq = (const float*)d_in[6],  *bq = (const float*)d_in[7];
    const float* wk = (const float*)d_in[8],  *bk = (const float*)d_in[9];
    const float* wv = (const float*)d_in[10], *bv = (const float*)d_in[11];
    const float* wo = (const float*)d_in[12], *bo = (const float*)d_in[13];
    const float* w1 = (const float*)d_in[14], *b1 = (const float*)d_in[15];
    const float* w2 = (const float*)d_in[16], *b2 = (const float*)d_in[17];
    const float* gen_w1 = (const float*)d_in[18];
    const float* gen_b1 = (const float*)d_in[19];
    const float* gen_lng = (const float*)d_in[20];
    const float* gen_lnb = (const float*)d_in[21];
    const float* gen_w2 = (const float*)d_in[22];
    const float* gen_b2 = (const float*)d_in[23];
    const float* gen_t  = (const float*)d_in[24];
    const float* dln_g  = (const float*)d_in[25];
    const float* dln_b  = (const float*)d_in[26];
    const float* wcat   = (const float*)d_in[27];
    const float* bcat   = (const float*)d_in[28];

    float* xout = (float*)d_out;                               // [S, H]
    float* wout = xout + (long long)S_LEN * H_DIM;             // [12, S, S]

    // ---------------- workspace (phase-reused) ----------------
    float* ws = (float*)d_ws;
    const long long SH = (long long)S_LEN * H_DIM;
    const long long NH = (long long)N_NODES * H_DIM;
    // Phase A
    float* xn  = ws;
    float* qb  = ws + 1 * SH;
    float* kb  = ws + 2 * SH;
    float* vb  = ws + 3 * SH;
    float* ctx = ws + 4 * SH;
    float* x1  = ws + 5 * SH;
    float* xf  = ws + 6 * SH;
    float* mb  = ws + 7 * SH;                                  // [S, 3072]
    // Phase B (reuses phase A memory; phase A is dead by then)
    unsigned* keys = (unsigned*)ws;                            // [N, H]
    float* den   = ws + 1 * NH;
    float* num   = ws + 2 * NH;
    float* agg   = ws + 3 * NH;
    float* tmp2h = ws + 4 * NH;                                // [N, 1536]
    float* convt = ws + 6 * NH;                                // [N, 768]
    float* gcat  = ws + 7 * NH;                                // [N, 2304]
    float* gfin  = ws + 10 * NH;                               // [N, 768]

    const dim3 blk(256);

    // ================= Transformer block =================
    ln_kernel<<<S_LEN, blk, 0, stream>>>(x, H_DIM, ln1_g, ln1_b, nullptr, 0,
                                         xn, H_DIM, H_DIM, 1e-6f, 0);
    gemm_f16_wmma<<<gemm_grid(S_LEN, H_DIM, 1), blk, 0, stream>>>(
        xn, 0, H_DIM, wq, 0, H_DIM, 0, bq, nullptr, 0, 0,
        qb, 0, H_DIM, S_LEN, H_DIM, H_DIM, 1.0f, 0);
    gemm_f16_wmma<<<gemm_grid(S_LEN, H_DIM, 1), blk, 0, stream>>>(
        xn, 0, H_DIM, wk, 0, H_DIM, 0, bk, nullptr, 0, 0,
        kb, 0, H_DIM, S_LEN, H_DIM, H_DIM, 1.0f, 0);
    gemm_f16_wmma<<<gemm_grid(S_LEN, H_DIM, 1), blk, 0, stream>>>(
        xn, 0, H_DIM, wv, 0, H_DIM, 0, bv, nullptr, 0, 0,
        vb, 0, H_DIM, S_LEN, H_DIM, H_DIM, 1.0f, 0);
    // scores[h] = (q_h @ k_h^T) / 8 -> directly into weights output
    gemm_f16_wmma<<<gemm_grid(S_LEN, S_LEN, HEADS), blk, 0, stream>>>(
        qb, DH, H_DIM, kb, DH, H_DIM, /*bT=*/1, nullptr, nullptr, 0, 0,
        wout, (long long)S_LEN * S_LEN, S_LEN,
        S_LEN, S_LEN, DH, 0.125f, 0);
    // softmax in place (this IS the second reference output)
    softmax_rows<<<HEADS * S_LEN, blk, 0, stream>>>(wout, S_LEN);
    // ctx[h] = weights_h @ v_h
    gemm_f16_wmma<<<gemm_grid(S_LEN, DH, HEADS), blk, 0, stream>>>(
        wout, (long long)S_LEN * S_LEN, S_LEN, vb, DH, H_DIM, /*bT=*/0,
        nullptr, nullptr, 0, 0,
        ctx, DH, H_DIM, S_LEN, DH, S_LEN, 1.0f, 0);
    // x1 = ctx @ wo + bo + x
    gemm_f16_wmma<<<gemm_grid(S_LEN, H_DIM, 1), blk, 0, stream>>>(
        ctx, 0, H_DIM, wo, 0, H_DIM, 0, bo, x, 0, H_DIM,
        x1, 0, H_DIM, S_LEN, H_DIM, H_DIM, 1.0f, 0);
    ln_kernel<<<S_LEN, blk, 0, stream>>>(x1, H_DIM, ln2_g, ln2_b, nullptr, 0,
                                         xf, H_DIM, H_DIM, 1e-6f, 0);
    // mb = gelu(xf @ w1 + b1)
    gemm_f16_wmma<<<gemm_grid(S_LEN, MLP_DIM, 1), blk, 0, stream>>>(
        xf, 0, H_DIM, w1, 0, MLP_DIM, 0, b1, nullptr, 0, 0,
        mb, 0, MLP_DIM, S_LEN, MLP_DIM, H_DIM, 1.0f, /*gelu*/1);
    // xout = mb @ w2 + b2 + x1   (final transformer x, first reference output)
    gemm_f16_wmma<<<gemm_grid(S_LEN, H_DIM, 1), blk, 0, stream>>>(
        mb, 0, MLP_DIM, w2, 0, H_DIM, 0, b2, x1, 0, H_DIM,
        xout, 0, H_DIM, S_LEN, H_DIM, MLP_DIM, 1.0f, 0);

    // ================= Part_GCN on node tokens =================
    const float* nodes = xout + H_DIM;       // x[0, 1:, :], ld = H
    const int egrid = (E_EDGES * H_DIM + 255) / 256;
    const int ngrid = (int)((NH + 255) / 256);

    for (int i = 0; i < 3; ++i) {
        const float* xin;
        int ldin;
        if (i == 0)      { xin = nodes;           ldin = H_DIM;  }   // x nodes
        else if (i == 1) { xin = gcat;            ldin = H3_DIM; }   // g0 (slice 0)
        else             { xin = gcat + H_DIM;    ldin = H3_DIM; }   // g1 (slice 1)

        hipMemsetAsync(keys, 0, (size_t)(3 * NH) * sizeof(float), stream);
        edge_max_kernel<<<egrid, blk, 0, stream>>>(xin, ldin, src, dst, keys, gen_t, i);
        edge_sum_kernel<<<egrid, blk, 0, stream>>>(xin, ldin, src, dst, keys, den, num, gen_t, i);
        agg_kernel<<<ngrid, blk, 0, stream>>>(num, den, xin, ldin, agg);

        // tmp2h = agg @ gen_w1[i] + gen_b1[i]
        gemm_f16_wmma<<<gemm_grid(N_NODES, H2_DIM, 1), blk, 0, stream>>>(
            agg, 0, H_DIM, gen_w1 + (long long)i * H_DIM * H2_DIM, 0, H2_DIM, 0,
            gen_b1 + i * H2_DIM, nullptr, 0, 0,
            tmp2h, 0, H2_DIM, N_NODES, H2_DIM, H_DIM, 1.0f, 0);
        // in-place: tmp2h = relu(LN(tmp2h))
        ln_kernel<<<N_NODES, blk, 0, stream>>>(tmp2h, H2_DIM,
            gen_lng + i * H2_DIM, gen_lnb + i * H2_DIM, nullptr, 0,
            tmp2h, H2_DIM, H2_DIM, 1e-5f, /*relu*/1);
        // conv output = tmp2h @ gen_w2[i] + gen_b2[i]
        float* convdst = (i == 0) ? gcat : convt;   // g0 goes straight to slice 0
        int ldconv = (i == 0) ? H3_DIM : H_DIM;
        gemm_f16_wmma<<<gemm_grid(N_NODES, H_DIM, 1), blk, 0, stream>>>(
            tmp2h, 0, H2_DIM, gen_w2 + (long long)i * H2_DIM * H_DIM, 0, H_DIM, 0,
            gen_b2 + i * H_DIM, nullptr, 0, 0,
            convdst, 0, ldconv, N_NODES, H_DIM, H2_DIM, 1.0f, 0);

        if (i == 1) {
            // g1 = g0 + relu(LN(conv, dln[0]))  -> gcat slice 1
            ln_kernel<<<N_NODES, blk, 0, stream>>>(convt, H_DIM,
                dln_g, dln_b, /*res=*/gcat, H3_DIM,
                gcat + H_DIM, H3_DIM, H_DIM, 1e-5f, /*relu*/1);
        } else if (i == 2) {
            // g2 = g1 + relu(LN(conv, dln[1]))  -> gcat slice 2
            ln_kernel<<<N_NODES, blk, 0, stream>>>(convt, H_DIM,
                dln_g + H_DIM, dln_b + H_DIM, /*res=*/gcat + H_DIM, H3_DIM,
                gcat + 2 * H_DIM, H3_DIM, H_DIM, 1e-5f, /*relu*/1);
        }
    }

    // g = concat(g0,g1,g2) @ wcat + bcat
    gemm_f16_wmma<<<gemm_grid(N_NODES, H_DIM, 1), blk, 0, stream>>>(
        gcat, 0, H3_DIM, wcat, 0, H_DIM, 0, bcat, nullptr, 0, 0,
        gfin, 0, H_DIM, N_NODES, H_DIM, H3_DIM, 1.0f, 0);

    // x[:,0,:] += mean(g, axis=0)
    mean_add_kernel<<<H_DIM, blk, 0, stream>>>(gfin, xout);
}
